// NNMFLayer_86560770883926
// MI455X (gfx1250) — compile-verified, matrix operand order checked by reading, not run on hardware
//
#include <hip/hip_runtime.h>

// ---------------------------------------------------------------------------
// NNMF layer forward for MI455X (gfx1250, wave32, WMMA 16x16x32 f16).
//
// Shapes: x[16,64,64,64] f32, W[1600,96] f32 (col-normalized), out h[16,96,60,60].
// Fused multiplicative-update NMF: per spatial tile everything (denom, ratio,
// bracket, h update, renorm) stays in LDS/registers; only x, W-fragments and
// the final h touch global memory.
// ---------------------------------------------------------------------------

typedef _Float16 v16h  __attribute__((ext_vector_type(16)));
typedef __fp16   v2fp  __attribute__((ext_vector_type(2)));
typedef float    v8f   __attribute__((ext_vector_type(8)));

#define D_TOT   1600
#define N_NEUR  96
#define KHW     25
#define EPSF    1e-20f

union Frag {
    v16h v;
    _Float16 h[16];
    unsigned int u[8];
};

__device__ __forceinline__ v8f vzero8() {
    v8f z;
#pragma unroll
    for (int i = 0; i < 8; ++i) z[i] = 0.f;
    return z;
}

// K index inside a 16-bit A/B WMMA fragment for vgpr slot v (0..7),
// lane-half hi (0: lanes 0-15, 1: lanes 16-31), pair element e (0/1).
// Matches ISA 16-bit A 16x32 table: v0 -> K{0,1}/{8,9}, v4 -> K{16,17}/{24,25}.
__device__ __forceinline__ int klocal(int v, int hi, int e) {
    int base = (v < 4) ? (2 * v + 8 * hi) : (16 + 2 * (v - 4) + 8 * hi);
    return base + e;
}

__device__ __forceinline__ v8f wmma_f16(v16h a, v16h b, v8f c) {
    return __builtin_amdgcn_wmma_f32_16x16x32_f16(
        /*neg_a=*/false, a, /*neg_b=*/false, b,
        /*c_mod=*/(short)0, c, /*reuse_a=*/false, /*reuse_b=*/false);
}

// ---------------------------------------------------------------------------
// One-time W fragment pre-swizzle into workspace (f32 -> f16, per-lane layout).
// b1f: GEMM1 B = W^T chunks  [32n x 16d], 300 frags of 32 lanes x 16 f16.
// b2f: GEMM2 B = W   chunks  [32d x 16n], 300 frags of 32 lanes x 16 f16.
// Each fragment lane then loads 16 contiguous f16 (32B) in the main kernel.
// ---------------------------------------------------------------------------
__global__ void nnmf_prep_wfrags(const float* __restrict__ W,
                                 _Float16* __restrict__ b1f,
                                 _Float16* __restrict__ b2f) {
    int idx = blockIdx.x * 256 + threadIdx.x;     // 0 .. 153599
    if (idx >= 300 * 32 * 16) return;
    int s    = idx & 15;          // slot within lane's 16 f16
    int lane = (idx >> 4) & 31;
    int fid  = idx >> 9;          // fragment id 0..299
    int v  = s >> 1, e = s & 1, hi = lane >> 4, li = lane & 15;

    // GEMM1: frag (kc, dt): B[k=n][col=d], n = kc*32 + klocal, d = dt*16 + li
    int kc = fid / 100, dt = fid % 100;
    int n1 = kc * 32 + klocal(v, hi, e);
    int d1 = dt * 16 + li;
    b1f[idx] = (_Float16)W[d1 * N_NEUR + n1];

    // GEMM2: frag (db, nb): B[k=d][col=n], d = db*32 + klocal, n = nb*16 + li
    int db = fid / 6, nb = fid % 6;
    int n2 = nb * 16 + li;
    int d2 = db * 32 + klocal(v, hi, e);
    b2f[idx] = (_Float16)W[d2 * N_NEUR + n2];
}

// ---------------------------------------------------------------------------
// Main fused kernel. Grid: (60 tiles, 16 batches), block: 128 threads (4 waves).
// Tile = 4 output rows x 16 output cols; wave wv owns output row (rt*4 + wv).
// Dynamic LDS layout (bytes):
//   xf   [64ch][8r][20c] f32 : 0      .. 40960
//   h    [64p][96n] f32      : 40960  .. 65536
//   Sp   [128] f32           : 65536  .. 66048
//   invS [64] f32            : 66048  .. 66304
//   stg  [4wv][16p][32d] f16 : 66304  .. 70400
// ~70 KB/WG -> up to 4 workgroups per 320 KB WGP.
// ---------------------------------------------------------------------------
__global__ void __launch_bounds__(128)
nnmf_forward(const float* __restrict__ x,
             const _Float16* __restrict__ b1f,
             const _Float16* __restrict__ b2f,
             float* __restrict__ out) {
    extern __shared__ char smem_raw[];
    float*     xf   = (float*)(smem_raw);
    float*     h_s  = (float*)(smem_raw + 40960);
    float*     Sp   = (float*)(smem_raw + 65536);
    float*     invS = (float*)(smem_raw + 66048);
    _Float16*  stg  = (_Float16*)(smem_raw + 66304);

    const int tid = threadIdx.x;
    const int b   = blockIdx.y;
    const int rt  = blockIdx.x >> 2;       // 0..14 row tile
    const int ct  = blockIdx.x & 3;        // 0..3  col tile
    const int yb  = rt * 4;
    const int xb  = ct * 16;

    // ---- Phase 0: stage input footprint (20x8 per channel), init h ----
    for (int idx = tid; idx < 64 * 8 * 20; idx += 128) {
        int c = idx / 160, rem = idx % 160;
        int ry = rem / 20, cx = rem % 20;
        int gx = xb + cx;
        float v = 0.f;
        if (gx < 64) v = x[((b * 64 + c) * 64 + (yb + ry)) * 64 + gx];
        xf[idx] = v;
    }
    for (int idx = tid; idx < 64 * N_NEUR; idx += 128)
        h_s[idx] = 1.0f / 96.0f;
    __syncthreads();

    // ---- per-position patch sum S[p] (for X normalization) ----
    {
        int p = tid & 63, half = tid >> 6;
        int pr = p >> 4, pc = p & 15;
        float s = 0.f;
        for (int c = half * 32; c < half * 32 + 32; ++c)
#pragma unroll
            for (int i = 0; i < 5; ++i) {
                int base = (c * 8 + pr + i) * 20 + pc;
#pragma unroll
                for (int j = 0; j < 5; ++j) s += xf[base + j];
            }
        Sp[tid] = s;
    }
    __syncthreads();
    if (tid < 64) invS[tid] = __builtin_amdgcn_rcpf(Sp[tid] + Sp[tid + 64] + EPSF);
    __syncthreads();

    // ---- per-wave setup ----
    const int lane = tid & 31;
    const int wv   = tid >> 5;       // 0..3: output row within tile, M-tile id
    const int li   = lane & 15;
    const int hi   = lane >> 4;
    const int hi8  = hi * 8;
    const int pbase = wv * 16;
    _Float16* stgw = stg + wv * (16 * 32);

    float invS_r[8];
#pragma unroll
    for (int r = 0; r < 8; ++r) invS_r[r] = invS[pbase + r + hi8];

    // Initial (db=0) decomposition of this lane's two d-columns into
    // (channel, kernel-row, kernel-col) and the resulting xf base offset.
    // d advances by 32 per D-block; maintained incrementally in the hot loop.
    int ijA0, jA0, xoA0, ijB0, jB0, xoB0;
    {
        int ij = li;                      // d0 = li < 25 -> c = 0
        int i = ij / 5, j = ij - i * 5;
        ijA0 = ij; jA0 = j;
        xoA0 = (wv + i) * 20 + j;
        int d1 = li + 16;
        int c = d1 / 25; ij = d1 - c * 25;
        i = ij / 5; j = ij - i * 5;
        ijB0 = ij; jB0 = j;
        xoB0 = (c * 8 + wv + i) * 20 + j;
    }

    Frag hA[3];
    v8f  acc2[6];

    // ---- 5 multiplicative-update iterations, fully wave-local ----
    for (int it = 0; it < 5; ++it) {
        // Build A-fragments of h (16p x 96n f16) from LDS fp32 h.
#pragma unroll
        for (int kc = 0; kc < 3; ++kc)
#pragma unroll
            for (int s = 0; s < 16; ++s) {
                int v = s >> 1, e = s & 1;
                int n = kc * 32 + klocal(v, hi, e);
                hA[kc].h[s] = (_Float16)h_s[(pbase + li) * N_NEUR + n];
            }
#pragma unroll
        for (int nb = 0; nb < 6; ++nb) acc2[nb] = vzero8();

        int ija = ijA0, ja = jA0, xoa = xoA0;
        int ijb = ijB0, jb = jB0, xob = xoB0;

#pragma unroll 1
        for (int db = 0; db < 50; ++db) {          // D in blocks of 32
            __builtin_prefetch(b2f + ((size_t)(db * 6) * 32) * 16, 0, 1);

            // GEMM1: denom[16p x 32d] = h[16p x 96n] @ W^T[96n x 32d]
            v8f a1 = vzero8(), a1b = vzero8();
#pragma unroll
            for (int kc = 0; kc < 3; ++kc) {
                v16h ba = *(const v16h*)(b1f + ((size_t)((kc * 100 + db * 2 + 0) * 32 + lane)) * 16);
                v16h bb = *(const v16h*)(b1f + ((size_t)((kc * 100 + db * 2 + 1) * 32 + lane)) * 16);
                a1  = wmma_f16(hA[kc].v, ba, a1);
                a1b = wmma_f16(hA[kc].v, bb, a1b);
            }

            // ratio = X/(denom+eps): X recomputed from input footprint in LDS.
            // This lane covers d columns db*32+li and db*32+li+16 (D-layout N dim).
#pragma unroll
            for (int r = 0; r < 8; ++r) {
                int pl = r + hi8;                        // D-layout M row
                float ra = xf[xoa + pl] * invS_r[r] * __builtin_amdgcn_rcpf(a1[r]  + EPSF);
                float rb = xf[xob + pl] * invS_r[r] * __builtin_amdgcn_rcpf(a1b[r] + EPSF);
                v2fp pk = __builtin_amdgcn_cvt_pkrtz(ra, rb);   // v_cvt_pk_rtz_f16_f32
                unsigned int bits = __builtin_bit_cast(unsigned int, pk);
                stgw[pl * 32 + li] =
                    __builtin_bit_cast(_Float16, (unsigned short)(bits & 0xffffu));
                stgw[pl * 32 + li + 16] =
                    __builtin_bit_cast(_Float16, (unsigned short)(bits >> 16));
            }
            // Same-wave DS ops are processed in order (HW RAW safe); this is a
            // compiler-only fence so the u32 reloads can't be hoisted above the
            // differently-typed f16 stores. HW waits are inserted by compiler.
            asm volatile("" ::: "memory");

            // Reload ratio as A-fragment [16p x 32d] (pairs are contiguous d).
            Frag rA;
#pragma unroll
            for (int v = 0; v < 8; ++v) {
                int k0 = klocal(v, hi, 0);
                rA.u[v] = *(const unsigned int*)&stgw[li * 32 + k0];
            }

            // GEMM2: bracket[16p x 96n] += ratio[16p x 32d] @ W[32d x 96n]
#pragma unroll
            for (int nb = 0; nb < 6; ++nb) {
                v16h b2v = *(const v16h*)(b2f + ((size_t)((db * 6 + nb) * 32 + lane)) * 16);
                acc2[nb] = wmma_f16(rA.v, b2v, acc2[nb]);
            }

            // d += 32: incremental (c,i,j)->xf-offset update, no divisions.
            // 32 = 25 + 7: ij += 7 (c += 1), j += 2 (i += 1); then carries.
            ija += 7; ja += 2; xoa += 182;               // +160(c) +20(i) +2(j)
            if (ja >= 5)   { ja -= 5;   xoa += 15; }     // j-carry: +20 - 5
            if (ija >= 25) { ija -= 25; xoa += 60; }     // c-carry: +160 - 100
            ijb += 7; jb += 2; xob += 182;
            if (jb >= 5)   { jb -= 5;   xob += 15; }
            if (ijb >= 25) { ijb -= 25; xob += 60; }
        }

        // h <- h * (1 + bracket)   (EPS0 = 1.0), in-place in LDS (wave-local rows)
#pragma unroll
        for (int nb = 0; nb < 6; ++nb)
#pragma unroll
            for (int r = 0; r < 8; ++r) {
                int pl = r + hi8;
                int n  = nb * 16 + li;
                float hv = h_s[(pbase + pl) * N_NEUR + n];
                h_s[(pbase + pl) * N_NEUR + n] = hv * (1.0f + acc2[nb][r]);
            }
        asm volatile("" ::: "memory");

        // renormalize over n per position: 2 lanes per position, shfl-combine
        {
            float s = 0.f;
            int base = (pbase + li) * N_NEUR + hi * 48;
#pragma unroll
            for (int n = 0; n < 48; ++n) s += h_s[base + n];
            s += __shfl_xor(s, 16, 32);
            float inv = __builtin_amdgcn_rcpf(s + EPSF);
#pragma unroll
            for (int n = 0; n < 48; ++n) h_s[base + n] *= inv;
        }
        asm volatile("" ::: "memory");
    }

    // ---- store h tile, coalesced over x, guard padded columns ----
    __syncthreads();
    for (int idx = tid; idx < 64 * N_NEUR; idx += 128) {
        int n = idx >> 6, p = idx & 63;
        int pr = p >> 4, pc = p & 15;
        int xcol = xb + pc;
        if (xcol < 60)
            out[((b * N_NEUR + n) * 60 + (yb + pr)) * 60 + xcol] = h_s[p * N_NEUR + n];
    }
}

// ---------------------------------------------------------------------------
extern "C" void kernel_launch(void* const* d_in, const int* in_sizes, int n_in,
                              void* d_out, int out_size, void* d_ws, size_t ws_size,
                              hipStream_t stream) {
    const float* x = (const float*)d_in[0];   // [16,64,64,64]
    const float* w = (const float*)d_in[1];   // [1600,96]
    float* out = (float*)d_out;               // [16,96,60,60]

    // workspace: 2 x 300 fragments x 32 lanes x 16 f16 = 614400 bytes
    _Float16* b1f = (_Float16*)d_ws;
    _Float16* b2f = b1f + 300 * 32 * 16;

    nnmf_prep_wfrags<<<600, 256, 0, stream>>>(w, b1f, b2f);

    const size_t smem = 70400;                // 40960 + 24576 + 512 + 256 + 4096
    nnmf_forward<<<dim3(60, 16), 128, smem, stream>>>(x, b1f, b2f, out);
}
